// TransformerLayer_9577777070537
// MI455X (gfx1250) — matrix operand order repaired and checked by reading, not compiled, not run on hardware
//
#include <hip/hip_runtime.h>

// ---------------------------------------------------------------------------
// TransformerLayer on MI455X (gfx1250): bf16 WMMA everywhere, flash attention,
// TDM (tensor_load_to_lds) double-buffered staging in the FFN1 GEMM.
// ---------------------------------------------------------------------------

#define B_ 2
#define S_ 2048
#define D_ 1024
#define H_ 16
#define E_ 64
#define F_ 4096

typedef __bf16 v16bf __attribute__((ext_vector_type(16)));
typedef float  v8f   __attribute__((ext_vector_type(8)));
typedef unsigned int u32x4 __attribute__((ext_vector_type(4)));
typedef int i32x8 __attribute__((ext_vector_type(8)));
typedef int i32x4 __attribute__((ext_vector_type(4)));

#if defined(__gfx1250__) && __has_builtin(__builtin_amdgcn_tensor_load_to_lds) && \
    __has_builtin(__builtin_amdgcn_s_wait_tensorcnt)
#define HAVE_TDM 1
#else
#define HAVE_TDM 0
#endif

union ABfrag { v16bf v; u32x4 q[2]; };

// ---- WMMA wrapper ----------------------------------------------------------
__device__ __forceinline__ v8f wmma_bf16(v16bf a, v16bf b, v8f c) {
  return __builtin_amdgcn_wmma_f32_16x16x32_bf16(false, a, false, b,
                                                 (short)0, c, false, false);
}

// ---- A fragment (16x32, row-major source, lda in elements) -----------------
// lane&15 = row; lane>>4 selects K half: 16 contiguous K per lane.
__device__ __forceinline__ v16bf load_a_g(const __bf16* __restrict__ base,
                                          int lda, int lane) {
  int m = lane & 15, hl = lane >> 4;
  const __bf16* r = base + (size_t)m * lda + hl * 16;
  ABfrag f;
  f.q[0] = *(const u32x4*)(r);        // K = hl*16 .. +7
  f.q[1] = *(const u32x4*)(r + 8);    // K = hl*16 + 8 .. +7
  return f.v;
}

// ---- B fragment (32x16) from pre-transposed BT[n][k] (ldb in elements) -----
// lane&15 = column n; lane>>4 selects K half: 16 contiguous K per lane.
__device__ __forceinline__ v16bf load_b_g(const __bf16* __restrict__ base,
                                          int ldb, int lane) {
  int n = lane & 15, hl = lane >> 4;
  const __bf16* r = base + (size_t)n * ldb + hl * 16;
  ABfrag f;
  f.q[0] = *(const u32x4*)(r);
  f.q[1] = *(const u32x4*)(r + 8);
  return f.v;
}

#if HAVE_TDM
// ---- TDM: load a 2-D tile [tile_rows x tile_k] (bf16) into LDS -------------
// Source is row-major [tensor_rows x tensor_k], leading dim = ld elements.
// D# layout per CDNA5 ISA ch.8 (group0: addr/count/type, group1: dims/strides)
// This toolchain exposes the 6-arg builtin:
//   (u32x4 g0, i32x8 g1, i32x4 g2, i32x4 g3, i32x8 g4, i32 cpol)
__device__ __forceinline__ void tdm_load_tile(unsigned lds_addr,
                                              const __bf16* g,
                                              unsigned tile_k, unsigned tile_rows,
                                              unsigned tensor_k, unsigned tensor_rows,
                                              unsigned ld) {
  unsigned long long ga = (unsigned long long)(uintptr_t)g;
  u32x4 g0;
  g0[0] = 1u;                                          // count=1, user mode
  g0[1] = lds_addr;                                    // LDS byte address
  g0[2] = (unsigned)(ga & 0xFFFFFFFFull);              // global_addr[31:0]
  g0[3] = ((unsigned)((ga >> 32) & 0x01FFFFFFull))     // global_addr[56:32]
          | 0x80000000u;                               // type=2 ("image")
  i32x8 g1;
  g1[0] = (int)(1u << 16);                             // data_size=1 (2 bytes)
  g1[1] = (int)((tensor_k & 0xFFFFu) << 16);           // tensor_dim0[15:0]
  g1[2] = (int)((tensor_k >> 16) | ((tensor_rows & 0xFFFFu) << 16));
  g1[3] = (int)((tensor_rows >> 16) | (tile_k << 16)); // tile_dim0
  g1[4] = (int)tile_rows;                              // tile_dim1
  g1[5] = (int)ld;                                     // tensor_dim0_stride[31:0]
  g1[6] = 0;
  g1[7] = 0;
  i32x4 gz4 = {0, 0, 0, 0};                            // 2-D: groups 2/3 unused
  i32x8 gz8 = {0, 0, 0, 0, 0, 0, 0, 0};
  __builtin_amdgcn_tensor_load_to_lds(g0, g1, gz4, gz4, gz8, 0);
}
#endif

// ---------------------------------------------------------------------------
// Staging kernels: fp32 -> bf16 (+ transposes)
// ---------------------------------------------------------------------------
__global__ void k_cvt_x(const float* __restrict__ x, __bf16* __restrict__ xb,
                        __bf16* __restrict__ xT, int total) {
  for (int i = blockIdx.x * blockDim.x + threadIdx.x; i < total;
       i += gridDim.x * blockDim.x) {
    int b = i / (S_ * D_);
    int rem = i - b * (S_ * D_);
    int s = rem / D_;
    int d = rem - s * D_;
    float v = x[i];
    xb[i] = (__bf16)v;
    xT[((size_t)b * D_ + d) * S_ + s] = (__bf16)v;
  }
}

// dst[batch][c][r] = (bf16) src[batch][r][c]
__global__ void k_transpose(const float* __restrict__ src,
                            __bf16* __restrict__ dst, int R, int C, int total) {
  for (int i = blockIdx.x * blockDim.x + threadIdx.x; i < total;
       i += gridDim.x * blockDim.x) {
    int bb = i / (R * C);
    int rem = i - bb * (R * C);
    int r = rem / C;
    int c = rem - r * C;
    dst[((size_t)bb * C + c) * R + r] = (__bf16)src[i];
  }
}

// ---------------------------------------------------------------------------
// QK projection: q = (x Wq + bq)/8, k = x Wk + bk  -> bf16 [B,H,S,E]
// ---------------------------------------------------------------------------
__global__ __launch_bounds__(256)
void k_qk(const __bf16* __restrict__ xb, const __bf16* __restrict__ wqT,
          const __bf16* __restrict__ wkT, const float* __restrict__ bq,
          const float* __restrict__ bk, __bf16* __restrict__ qo,
          __bf16* __restrict__ ko) {
  int lane = threadIdx.x & 31, wave = threadIdx.x >> 5;
  int wid = blockIdx.x * 8 + wave;           // 16 heads * 128 mtiles * 2 etiles
  int et = wid & 1;
  int mt = (wid >> 1) & 127;
  int h  = wid >> 8;
  int m0 = mt * 32, e0 = et * 32;
  const __bf16* A0 = xb + (size_t)m0 * D_;
  const __bf16* A1 = xb + (size_t)(m0 + 16) * D_;
  const __bf16* Bq = wqT + ((size_t)h * E_ + e0) * D_;
  const __bf16* Bk = wkT + ((size_t)h * E_ + e0) * D_;
  v8f z = {0.f,0.f,0.f,0.f,0.f,0.f,0.f,0.f};
  v8f cq[2][2], ck[2][2];
  for (int i = 0; i < 2; ++i)
    for (int j = 0; j < 2; ++j) { cq[i][j] = z; ck[i][j] = z; }
  for (int k0 = 0; k0 < D_; k0 += 32) {
    v16bf a0 = load_a_g(A0 + k0, D_, lane);
    v16bf a1 = load_a_g(A1 + k0, D_, lane);
    v16bf bq0 = load_b_g(Bq + k0, D_, lane);
    v16bf bq1 = load_b_g(Bq + (size_t)16 * D_ + k0, D_, lane);
    v16bf bk0 = load_b_g(Bk + k0, D_, lane);
    v16bf bk1 = load_b_g(Bk + (size_t)16 * D_ + k0, D_, lane);
    cq[0][0] = wmma_bf16(a0, bq0, cq[0][0]);
    cq[0][1] = wmma_bf16(a0, bq1, cq[0][1]);
    cq[1][0] = wmma_bf16(a1, bq0, cq[1][0]);
    cq[1][1] = wmma_bf16(a1, bq1, cq[1][1]);
    ck[0][0] = wmma_bf16(a0, bk0, ck[0][0]);
    ck[0][1] = wmma_bf16(a0, bk1, ck[0][1]);
    ck[1][0] = wmma_bf16(a1, bk0, ck[1][0]);
    ck[1][1] = wmma_bf16(a1, bk1, ck[1][1]);
  }
  int n = lane & 15, hl = lane >> 4;
  for (int rt = 0; rt < 2; ++rt)
    for (int ct = 0; ct < 2; ++ct)
      for (int r = 0; r < 8; ++r) {
        int row = m0 + rt * 16 + r + 8 * hl;
        int bb = row >> 11;                  // row / S_
        int s  = row & (S_ - 1);
        int e  = e0 + ct * 16 + n;
        size_t o = (((size_t)(bb * H_ + h)) * S_ + s) * E_ + e;
        qo[o] = (__bf16)((cq[rt][ct][r] + bq[h * E_ + e]) * 0.125f);
        ko[o] = (__bf16)(ck[rt][ct][r] + bk[h * E_ + e]);
      }
}

// ---------------------------------------------------------------------------
// Flash attention: ctx[b,h,s,:] = softmax_causal(q k^T) @ x
// ---------------------------------------------------------------------------
__global__ __launch_bounds__(512)
void k_attn(const __bf16* __restrict__ qb, const __bf16* __restrict__ kb,
            const __bf16* __restrict__ xT, __bf16* __restrict__ ctxb) {
  __shared__ __align__(16) __bf16 ldsp[16][16][48];  // per-wave p scratch
  const int lane = threadIdx.x & 31, wave = threadIdx.x >> 5;
  const int n = lane & 15, hl = lane >> 4;
  const int rt = wave & 1;
  const int d0 = (wave >> 1) * 128;
  const int blk = blockIdx.x;                // B * H * (S/32) = 2048
  const int sb = blk & 63;
  const int h  = (blk >> 6) & 15;
  const int b  = blk >> 10;
  const int s0 = sb * 32;

  const __bf16* qbase = qb + ((((size_t)(b * H_ + h)) * S_) + s0 + rt * 16) * E_;
  const __bf16* kbase = kb + (((size_t)(b * H_ + h)) * S_) * E_;
  const __bf16* xbase = xT + ((size_t)b * D_ + d0) * S_;

  v16bf aq0 = load_a_g(qbase, E_, lane);
  v16bf aq1 = load_a_g(qbase + 32, E_, lane);

  v8f z = {0.f,0.f,0.f,0.f,0.f,0.f,0.f,0.f};
  v8f acc[8];
  for (int c = 0; c < 8; ++c) acc[c] = z;
  float mrun[8], lrun[8];
  for (int r = 0; r < 8; ++r) { mrun[r] = -3.0e38f; lrun[r] = 0.f; }

  for (int t0 = 0; t0 <= s0; t0 += 32) {
    v8f sc[2];
    #pragma unroll
    for (int j = 0; j < 2; ++j) {
      v16bf bk0 = load_b_g(kbase + (size_t)(t0 + j * 16) * E_, E_, lane);
      v16bf bk1 = load_b_g(kbase + (size_t)(t0 + j * 16) * E_ + 32, E_, lane);
      v8f c = z;
      c = wmma_bf16(aq0, bk0, c);
      c = wmma_bf16(aq1, bk1, c);
      sc[j] = c;
    }
    if (t0 + 31 > s0) {
      #pragma unroll
      for (int j = 0; j < 2; ++j)
        #pragma unroll
        for (int r = 0; r < 8; ++r) {
          int sg = s0 + rt * 16 + r + 8 * hl;
          int tg = t0 + j * 16 + n;
          if (tg > sg) sc[j][r] = -1.0e9f;
        }
    }
    float pscale[8];
    #pragma unroll
    for (int r = 0; r < 8; ++r) {
      float v = fmaxf(sc[0][r], sc[1][r]);
      for (int m = 1; m < 16; m <<= 1) v = fmaxf(v, __shfl_xor(v, m, 32));
      float mnew = fmaxf(mrun[r], v);
      float so = __expf(mrun[r] - mnew);
      mrun[r] = mnew;
      float p0 = __expf(sc[0][r] - mnew);
      float p1 = __expf(sc[1][r] - mnew);
      sc[0][r] = p0; sc[1][r] = p1;
      float rs = p0 + p1;
      for (int m = 1; m < 16; m <<= 1) rs += __shfl_xor(rs, m, 32);
      lrun[r] = lrun[r] * so + rs;
      pscale[r] = so;
    }
    #pragma unroll
    for (int c = 0; c < 8; ++c)
      #pragma unroll
      for (int r = 0; r < 8; ++r) acc[c][r] *= pscale[r];

    #pragma unroll
    for (int j = 0; j < 2; ++j)
      #pragma unroll
      for (int r = 0; r < 8; ++r)
        ldsp[wave][r + 8 * hl][j * 16 + n] = (__bf16)sc[j][r];
    asm volatile("s_wait_dscnt 0" ::: "memory");
    v16bf pa;
    {
      const __bf16* pr = &ldsp[wave][n][hl * 16];
      ABfrag f;
      f.q[0] = *(const u32x4*)(pr);
      f.q[1] = *(const u32x4*)(pr + 8);
      pa = f.v;
    }
    #pragma unroll
    for (int c = 0; c < 8; ++c) {
      v16bf bx = load_b_g(xbase + (size_t)(c * 16) * S_ + t0, S_, lane);
      acc[c] = wmma_bf16(pa, bx, acc[c]);
    }
  }

  float inv[8];
  for (int r = 0; r < 8; ++r) inv[r] = 1.0f / lrun[r];
  __bf16* obase = ctxb + ((((size_t)(b * H_ + h)) * S_) + s0 + rt * 16) * D_;
  for (int c = 0; c < 8; ++c)
    for (int r = 0; r < 8; ++r)
      obase[(size_t)(r + 8 * hl) * D_ + d0 + c * 16 + n] =
          (__bf16)(acc[c][r] * inv[r]);
}

// ---------------------------------------------------------------------------
// attn_sum[b,s,d] = sum_h relu(ctx_h @ Wo_h + bo_h)
// ---------------------------------------------------------------------------
__global__ __launch_bounds__(256)
void k_headsum(const __bf16* __restrict__ ctxb, const __bf16* __restrict__ woT,
               const float* __restrict__ bo, float* __restrict__ asum) {
  int lane = threadIdx.x & 31, wave = threadIdx.x >> 5;
  int wid = blockIdx.x * 8 + wave;           // 128 mtiles * 32 ntiles
  int nt = wid & 31;
  int mt = wid >> 5;
  int m0 = mt * 32, o0 = nt * 32;
  int bb = m0 >> 11, s0 = m0 & (S_ - 1);
  int n = lane & 15, hl = lane >> 4;
  v8f z = {0.f,0.f,0.f,0.f,0.f,0.f,0.f,0.f};
  v8f sum[2][2];
  for (int i = 0; i < 2; ++i) for (int j = 0; j < 2; ++j) sum[i][j] = z;

  for (int h = 0; h < H_; ++h) {
    const __bf16* A  = ctxb + ((((size_t)(bb * H_ + h)) * S_) + s0) * D_;
    const __bf16* Bw = woT + ((size_t)h * D_ + o0) * D_;
    v8f c[2][2];
    for (int i = 0; i < 2; ++i) for (int j = 0; j < 2; ++j) c[i][j] = z;
    for (int k0 = 0; k0 < D_; k0 += 32) {
      v16bf a0 = load_a_g(A + k0, D_, lane);
      v16bf a1 = load_a_g(A + (size_t)16 * D_ + k0, D_, lane);
      v16bf b0 = load_b_g(Bw + k0, D_, lane);
      v16bf b1 = load_b_g(Bw + (size_t)16 * D_ + k0, D_, lane);
      c[0][0] = wmma_bf16(a0, b0, c[0][0]);
      c[0][1] = wmma_bf16(a0, b1, c[0][1]);
      c[1][0] = wmma_bf16(a1, b0, c[1][0]);
      c[1][1] = wmma_bf16(a1, b1, c[1][1]);
    }
    float bo0 = bo[h * D_ + o0 + n];
    float bo1 = bo[h * D_ + o0 + 16 + n];
    for (int rt = 0; rt < 2; ++rt)
      for (int r = 0; r < 8; ++r) {
        sum[rt][0][r] += fmaxf(c[rt][0][r] + bo0, 0.f);
        sum[rt][1][r] += fmaxf(c[rt][1][r] + bo1, 0.f);
      }
  }
  for (int rt = 0; rt < 2; ++rt)
    for (int ct = 0; ct < 2; ++ct)
      for (int r = 0; r < 8; ++r) {
        int row = m0 + rt * 16 + r + 8 * hl;
        asum[(size_t)row * D_ + o0 + ct * 16 + n] = sum[rt][ct][r];
      }
}

// ---------------------------------------------------------------------------
// LayerNorm 1: x1 = LN(x + attn_sum); writes fp32 + bf16 copies
// ---------------------------------------------------------------------------
__global__ __launch_bounds__(256)
void k_ln1(const float* __restrict__ x, const float* __restrict__ asum,
           const float* __restrict__ g, const float* __restrict__ be,
           float* __restrict__ x1f, __bf16* __restrict__ x1b) {
  __shared__ float red[512];
  __shared__ float mu_s, rs_s;
  int row = blockIdx.x, t = threadIdx.x;
  const float* xr = x + (size_t)row * D_;
  const float* ar = asum + (size_t)row * D_;
  float y[4], s1 = 0.f, s2 = 0.f;
  for (int i = 0; i < 4; ++i) {
    y[i] = xr[t * 4 + i] + ar[t * 4 + i];
    s1 += y[i];
    s2 += y[i] * y[i];
  }
  red[t] = s1; red[256 + t] = s2;
  __syncthreads();
  for (int o = 128; o > 0; o >>= 1) {
    if (t < o) { red[t] += red[t + o]; red[256 + t] += red[256 + t + o]; }
    __syncthreads();
  }
  if (t == 0) {
    float mu = red[0] * (1.0f / D_);
    float var = red[256] * (1.0f / D_) - mu * mu;
    mu_s = mu;
    rs_s = rsqrtf(var + 1e-3f);
  }
  __syncthreads();
  float mu = mu_s, rs = rs_s;
  for (int i = 0; i < 4; ++i) {
    int d = t * 4 + i;
    float o = (y[i] - mu) * rs * g[d] + be[d];
    x1f[(size_t)row * D_ + d] = o;
    x1b[(size_t)row * D_ + d] = (__bf16)o;
  }
}

// ---------------------------------------------------------------------------
// FFN1: hb = relu(x1 @ W1 + b1)  (bf16 out)
// Workgroup-cooperative 64x128 macro-tile; A/B tiles staged in LDS via TDM
// (tensor_load_to_lds, double-buffered, TENSORcnt-synchronized).
// ---------------------------------------------------------------------------
__global__ __launch_bounds__(256)
void k_ffn1(const __bf16* __restrict__ x1b, const __bf16* __restrict__ w1T,
            const float* __restrict__ b1, __bf16* __restrict__ hb) {
  __shared__ __align__(16) __bf16 ldsA[2][64][32];    //  8 KiB
  __shared__ __align__(16) __bf16 ldsB[2][128][32];   // 16 KiB
  const int lane = threadIdx.x & 31, wave = threadIdx.x >> 5;
  const int tid = threadIdx.x;
  const int bm = blockIdx.x >> 5;       // 64 m macro-tiles
  const int bn = blockIdx.x & 31;       // 32 n macro-tiles
  const int m_base = bm * 64, n_base = bn * 128;
  const int ry = wave >> 2, cx = wave & 3;
  const int m0 = m_base + ry * 32, n0 = n_base + cx * 32;
  const __bf16* Ag = x1b + (size_t)m_base * D_;
  const __bf16* Bg = w1T + (size_t)n_base * D_;
  v8f z = {0.f,0.f,0.f,0.f,0.f,0.f,0.f,0.f};
  v8f c[2][2];
  for (int i = 0; i < 2; ++i) for (int j = 0; j < 2; ++j) c[i][j] = z;

#if HAVE_TDM
  if (wave == 0) {                       // prologue: fill buffer 0
    tdm_load_tile((unsigned)(uintptr_t)&ldsA[0][0][0], Ag, 32, 64, D_, B_ * S_, D_);
    tdm_load_tile((unsigned)(uintptr_t)&ldsB[0][0][0], Bg, 32, 128, D_, F_, D_);
  }
#endif

  for (int it = 0; it < D_ / 32; ++it) {
    const int k0 = it * 32;
#if HAVE_TDM
    const int cur = it & 1, nxt = cur ^ 1;
    if (wave == 0) {
      if (it + 1 < D_ / 32) {
        tdm_load_tile((unsigned)(uintptr_t)&ldsA[nxt][0][0], Ag + k0 + 32,
                      32, 64, D_, B_ * S_, D_);
        tdm_load_tile((unsigned)(uintptr_t)&ldsB[nxt][0][0], Bg + k0 + 32,
                      32, 128, D_, F_, D_);
        __builtin_amdgcn_s_wait_tensorcnt(2);   // current buffer complete
      } else {
        __builtin_amdgcn_s_wait_tensorcnt(0);
      }
    }
    __syncthreads();
#else
    const int cur = 0;
    for (int i = tid; i < (64 * 32) / 8; i += 256) {
      int r = i >> 2, cc = i & 3;
      *(u32x4*)&ldsA[0][r][cc * 8] =
          *(const u32x4*)(Ag + (size_t)r * D_ + k0 + cc * 8);
    }
    for (int i = tid; i < (128 * 32) / 8; i += 256) {
      int r = i >> 2, cc = i & 3;
      *(u32x4*)&ldsB[0][r][cc * 8] =
          *(const u32x4*)(Bg + (size_t)r * D_ + k0 + cc * 8);
    }
    __syncthreads();
#endif
    v16bf a0  = load_a_g(&ldsA[cur][ry * 32 +  0][0], 32, lane);
    v16bf a1  = load_a_g(&ldsA[cur][ry * 32 + 16][0], 32, lane);
    v16bf b0  = load_b_g(&ldsB[cur][cx * 32 +  0][0], 32, lane);
    v16bf b1v = load_b_g(&ldsB[cur][cx * 32 + 16][0], 32, lane);
    c[0][0] = wmma_bf16(a0, b0,  c[0][0]);
    c[0][1] = wmma_bf16(a0, b1v, c[0][1]);
    c[1][0] = wmma_bf16(a1, b0,  c[1][0]);
    c[1][1] = wmma_bf16(a1, b1v, c[1][1]);
    __syncthreads();                     // buffer reuse fence
  }

  int n = lane & 15, hl = lane >> 4;
  for (int rt = 0; rt < 2; ++rt)
    for (int ct = 0; ct < 2; ++ct)
      for (int r = 0; r < 8; ++r) {
        int row = m0 + rt * 16 + r + 8 * hl;
        int col = n0 + ct * 16 + n;
        hb[(size_t)row * F_ + col] = (__bf16)fmaxf(c[rt][ct][r] + b1[col], 0.f);
      }
}

// ---------------------------------------------------------------------------
// FFN2 + residual: z = x1 + relu(hb @ W2 + b2)  (fp32 out)
// ---------------------------------------------------------------------------
__global__ __launch_bounds__(256)
void k_ffn2(const __bf16* __restrict__ hb, const __bf16* __restrict__ w2T,
            const float* __restrict__ b2, const float* __restrict__ x1f,
            float* __restrict__ zf) {
  int lane = threadIdx.x & 31, wave = threadIdx.x >> 5;
  int wid = blockIdx.x * 8 + wave;           // 128 mtiles * 32 ntiles
  int nt = wid & 31;
  int mt = wid >> 5;
  int m0 = mt * 32, n0 = nt * 32;
  const __bf16* A  = hb + (size_t)m0 * F_;
  const __bf16* Bw = w2T + (size_t)n0 * F_;
  v8f z = {0.f,0.f,0.f,0.f,0.f,0.f,0.f,0.f};
  v8f c[2][2];
  for (int i = 0; i < 2; ++i) for (int j = 0; j < 2; ++j) c[i][j] = z;
  for (int k0 = 0; k0 < F_; k0 += 32) {
    __builtin_prefetch(A + k0 + 256, 0, 0);
    v16bf a0 = load_a_g(A + k0, F_, lane);
    v16bf a1 = load_a_g(A + (size_t)16 * F_ + k0, F_, lane);
    v16bf b0 = load_b_g(Bw + k0, F_, lane);
    v16bf b1v = load_b_g(Bw + (size_t)16 * F_ + k0, F_, lane);
    c[0][0] = wmma_bf16(a0, b0,  c[0][0]);
    c[0][1] = wmma_bf16(a0, b1v, c[0][1]);
    c[1][0] = wmma_bf16(a1, b0,  c[1][0]);
    c[1][1] = wmma_bf16(a1, b1v, c[1][1]);
  }
  int n = lane & 15, hl = lane >> 4;
  for (int rt = 0; rt < 2; ++rt)
    for (int ct = 0; ct < 2; ++ct)
      for (int r = 0; r < 8; ++r) {
        int row = m0 + rt * 16 + r + 8 * hl;
        int col = n0 + ct * 16 + n;
        zf[(size_t)row * D_ + col] =
            x1f[(size_t)row * D_ + col] + fmaxf(c[rt][ct][r] + b2[col], 0.f);
      }
}

// ---------------------------------------------------------------------------
// LayerNorm 2: out = LN(z)
// ---------------------------------------------------------------------------
__global__ __launch_bounds__(256)
void k_ln2(const float* __restrict__ zf, const float* __restrict__ g,
           const float* __restrict__ be, float* __restrict__ out) {
  __shared__ float red[512];
  __shared__ float mu_s, rs_s;
  int row = blockIdx.x, t = threadIdx.x;
  const float* zr = zf + (size_t)row * D_;
  float y[4], s1 = 0.f, s2 = 0.f;
  for (int i = 0; i < 4; ++i) {
    y[i] = zr[t * 4 + i];
    s1 += y[i];
    s2 += y[i] * y[i];
  }
  red[t] = s1; red[256 + t] = s2;
  __syncthreads();
  for (int o = 128; o > 0; o >>= 1) {
    if (t < o) { red[t] += red[t + o]; red[256 + t] += red[256 + t + o]; }
    __syncthreads();
  }
  if (t == 0) {
    float mu = red[0] * (1.0f / D_);
    float var = red[256] * (1.0f / D_) - mu * mu;
    mu_s = mu;
    rs_s = rsqrtf(var + 1e-3f);
  }
  __syncthreads();
  float mu = mu_s, rs = rs_s;
  for (int i = 0; i < 4; ++i) {
    int d = t * 4 + i;
    out[(size_t)row * D_ + d] = (y[i] - mu) * rs * g[d] + be[d];
  }
}

// ---------------------------------------------------------------------------
// Workspace layout (bytes)
// ---------------------------------------------------------------------------
#define ALN(x) (((x) + 255) & ~(size_t)255)
static constexpr size_t SZ_XB  = (size_t)B_ * S_ * D_ * 2;   // 8 MiB
static constexpr size_t SZ_WQT = (size_t)H_ * E_ * D_ * 2;   // 2 MiB
static constexpr size_t SZ_WOT = (size_t)H_ * D_ * D_ * 2;   // 32 MiB
static constexpr size_t SZ_W1T = (size_t)D_ * F_ * 2;        // 8 MiB
static constexpr size_t SZ_QB  = (size_t)B_ * H_ * S_ * E_ * 2;
static constexpr size_t SZ_CTX = (size_t)B_ * H_ * S_ * D_ * 2; // 128 MiB
static constexpr size_t SZ_F32 = (size_t)B_ * S_ * D_ * 4;   // 16 MiB
static constexpr size_t SZ_HB  = (size_t)B_ * S_ * F_ * 2;   // 32 MiB

static constexpr size_t OFF_XB  = 0;
static constexpr size_t OFF_XT  = ALN(OFF_XB  + SZ_XB);
static constexpr size_t OFF_WQT = ALN(OFF_XT  + SZ_XB);
static constexpr size_t OFF_WKT = ALN(OFF_WQT + SZ_WQT);
static constexpr size_t OFF_WOT = ALN(OFF_WKT + SZ_WQT);
static constexpr size_t OFF_W1T = ALN(OFF_WOT + SZ_WOT);
static constexpr size_t OFF_W2T = ALN(OFF_W1T + SZ_W1T);
static constexpr size_t OFF_QB  = ALN(OFF_W2T + SZ_W1T);
static constexpr size_t OFF_KB  = ALN(OFF_QB  + SZ_QB);
static constexpr size_t OFF_CTX = ALN(OFF_KB  + SZ_QB);
static constexpr size_t OFF_AS  = ALN(OFF_CTX + SZ_CTX);
static constexpr size_t OFF_X1F = ALN(OFF_AS  + SZ_F32);
static constexpr size_t OFF_X1B = ALN(OFF_X1F + SZ_F32);
static constexpr size_t OFF_HB  = ALN(OFF_X1B + SZ_XB);
static constexpr size_t OFF_ZF  = ALN(OFF_HB  + SZ_HB);
static constexpr size_t WS_NEED = ALN(OFF_ZF  + SZ_F32);

extern "C" void kernel_launch(void* const* d_in, const int* in_sizes, int n_in,
                              void* d_out, int out_size, void* d_ws,
                              size_t ws_size, hipStream_t stream) {
  (void)in_sizes; (void)n_in; (void)out_size;
  if (ws_size < WS_NEED) return;

  const float* x    = (const float*)d_in[0];
  const float* Wq   = (const float*)d_in[1];
  const float* bq   = (const float*)d_in[2];
  const float* Wk   = (const float*)d_in[3];
  const float* bk   = (const float*)d_in[4];
  const float* Wo   = (const float*)d_in[5];
  const float* bo   = (const float*)d_in[6];
  const float* ln1g = (const float*)d_in[7];
  const float* ln1b = (const float*)d_in[8];
  const float* W1   = (const float*)d_in[9];
  const float* b1   = (const float*)d_in[10];
  const float* W2   = (const float*)d_in[11];
  const float* b2   = (const float*)d_in[12];
  const float* ln2g = (const float*)d_in[13];
  const float* ln2b = (const float*)d_in[14];

  char* ws = (char*)d_ws;
  __bf16* xb  = (__bf16*)(ws + OFF_XB);
  __bf16* xT  = (__bf16*)(ws + OFF_XT);
  __bf16* wqT = (__bf16*)(ws + OFF_WQT);
  __bf16* wkT = (__bf16*)(ws + OFF_WKT);
  __bf16* woT = (__bf16*)(ws + OFF_WOT);
  __bf16* w1T = (__bf16*)(ws + OFF_W1T);
  __bf16* w2T = (__bf16*)(ws + OFF_W2T);
  __bf16* qbw = (__bf16*)(ws + OFF_QB);
  __bf16* kbw = (__bf16*)(ws + OFF_KB);
  __bf16* ctx = (__bf16*)(ws + OFF_CTX);
  float*  asum = (float*)(ws + OFF_AS);
  float*  x1f  = (float*)(ws + OFF_X1F);
  __bf16* x1b  = (__bf16*)(ws + OFF_X1B);
  __bf16* hb   = (__bf16*)(ws + OFF_HB);
  float*  zf   = (float*)(ws + OFF_ZF);
  float*  out  = (float*)d_out;

  // ---- stage: fp32 -> bf16 (+ transposes) ----------------------------------
  k_cvt_x<<<8192, 256, 0, stream>>>(x, xb, xT, B_ * S_ * D_);
  k_transpose<<<2048, 256, 0, stream>>>(Wq, wqT, D_, E_, H_ * D_ * E_);
  k_transpose<<<2048, 256, 0, stream>>>(Wk, wkT, D_, E_, H_ * D_ * E_);
  k_transpose<<<8192, 256, 0, stream>>>(Wo, woT, D_, D_, H_ * D_ * D_);
  k_transpose<<<4096, 256, 0, stream>>>(W1, w1T, D_, F_, D_ * F_);
  k_transpose<<<4096, 256, 0, stream>>>(W2, w2T, F_, D_, F_ * D_);

  // ---- attention -----------------------------------------------------------
  k_qk<<<512, 256, 0, stream>>>(xb, wqT, wkT, bq, bk, qbw, kbw);
  k_attn<<<B_ * H_ * (S_ / 32), 512, 0, stream>>>(qbw, kbw, xT, ctx);
  k_headsum<<<512, 256, 0, stream>>>(ctx, woT, bo, asum);
  k_ln1<<<B_ * S_, 256, 0, stream>>>(x, asum, ln1g, ln1b, x1f, x1b);

  // ---- FFN -----------------------------------------------------------------
  k_ffn1<<<2048, 256, 0, stream>>>(x1b, w1T, b1, hb);
  k_ffn2<<<512, 256, 0, stream>>>(hb, w2T, b2, x1f, zf);
  k_ln2<<<B_ * S_, 256, 0, stream>>>(zf, ln2g, ln2b, out);
}
